// HeterogeneousGraphAttention_46866683134438
// MI455X (gfx1250) — matrix operand order repaired and checked by reading, not compiled, not run on hardware
//
#include <hip/hip_runtime.h>
#include <hip/hip_bf16.h>

// ---------------------------------------------------------------------------
// HeterogeneousGraphAttention for MI455X (gfx1250, wave32, WMMA + TDM)
//
//   1. wprep    : Wq/Wk/Wv/Wo f32[512,512] -> bf16 transposed [col][k]
//   2. maskpack : adjacency int32[B,N,N] -> bit mask u32[B,N,N/32]
//   3. qkv      : X @ W{q,k,v}; weight tiles staged to LDS by the Tensor
//                 Data Mover (tensor_load_to_lds, padded rows), WMMA bf16
//   4. attn     : flash attention, online softmax, prefetched K/V fragments,
//                 P relayout through LDS (ds_store/ds_load_b128)
//   5. proj     : attended @ Wo + bo + residual (same TDM staging)
//   6. ln       : per-row layernorm -> d_out
// ---------------------------------------------------------------------------

typedef __bf16 bf16;
typedef bf16  v16bf __attribute__((ext_vector_type(16)));
typedef bf16  v8bf  __attribute__((ext_vector_type(8)));
typedef float v8f   __attribute__((ext_vector_type(8)));
typedef float v4f   __attribute__((ext_vector_type(4)));
typedef unsigned tdm_g0 __attribute__((ext_vector_type(4)));
typedef unsigned tdm_g1 __attribute__((ext_vector_type(8)));

#define B_  4
#define N_  2048
#define D_  512
#define H_  8
#define HD_ 64
#define NW_ (N_ / 32)            // 64 mask words per row

// LDS weight tile: 64 rows x (256 bf16 data + 8 bf16 TDM pad) = 528B stride
#define WROW_ 264

// Workspace layout (bytes). Total = 0x3400000 (52 MiB).
#define OFF_WQT  ((size_t)0x0000000)
#define OFF_WKT  ((size_t)0x0080000)
#define OFF_WVT  ((size_t)0x0100000)
#define OFF_WOT  ((size_t)0x0180000)
#define OFF_Q    ((size_t)0x0200000)
#define OFF_K    ((size_t)0x0A00000)
#define OFF_VT   ((size_t)0x1200000)
#define OFF_MASK ((size_t)0x1A00000)
#define OFF_ATT  ((size_t)0x1C00000)
#define OFF_Y    ((size_t)0x2400000)

// ---------------------------------------------------------------------------
// WMMA helper: D = A(16x32 bf16) * B(32x16 bf16) + C(16x16 f32)
// ---------------------------------------------------------------------------
__device__ inline v8f wmma_bf16(v16bf a, v16bf b, v8f c) {
  return __builtin_amdgcn_wmma_f32_16x16x32_bf16(
      false, a, false, b, (short)0, c, false, false);
}

// A fragment (16x32 MxK) from an f32 row, converting to bf16.
__device__ inline v16bf load_a_f32(const float* __restrict__ row, int k0, int lane) {
  int kb = (lane & 16) ? 8 : 0;
  const float* p0 = row + k0 + kb;
  const float* p1 = p0 + 16;
  v4f x0 = *(const v4f*)(p0);
  v4f x1 = *(const v4f*)(p0 + 4);
  v4f x2 = *(const v4f*)(p1);
  v4f x3 = *(const v4f*)(p1 + 4);
  v16bf a;
#pragma unroll
  for (int e = 0; e < 4; ++e) {
    a[e]      = (bf16)x0[e];
    a[4 + e]  = (bf16)x1[e];
    a[8 + e]  = (bf16)x2[e];
    a[12 + e] = (bf16)x3[e];
  }
  return a;
}

// A fragment from a bf16 row.
__device__ inline v16bf load_a_bf16(const bf16* __restrict__ row, int k0, int lane) {
  int kb = (lane & 16) ? 8 : 0;
  union { v16bf v; v8bf h[2]; } u;
  u.h[0] = *(const v8bf*)(row + k0 + kb);
  u.h[1] = *(const v8bf*)(row + k0 + kb + 16);
  return u.v;
}

// B fragment (32x16 KxN); `colbase` = contiguous K-run for this lane's column.
__device__ inline v16bf load_b_bf16(const bf16* __restrict__ colbase, int k0, int lane) {
  int kb = (lane & 16) ? 8 : 0;
  union { v16bf v; v8bf h[2]; } u;
  u.h[0] = *(const v8bf*)(colbase + k0 + kb);
  u.h[1] = *(const v8bf*)(colbase + k0 + kb + 16);
  return u.v;
}

// ---------------------------------------------------------------------------
// TDM: stage a 64-row x 512-byte weight sub-tile into LDS.
//   2D tile, data_size = 8B units: tile_dim0 = 64 (512B/row),
//   tile_dim1 = 64 rows, row stride = 128 units (1024B),
//   pad_enable, pad_interval = 512B (code 6), pad_amount = 16B (code 3)
//   -> LDS rows land at 528B stride (WROW_ bf16).
// ---------------------------------------------------------------------------
__device__ inline void tdm_load_wtile(unsigned lds_off, const bf16* gsrc) {
  unsigned long long ga = (unsigned long long)(uintptr_t)gsrc;
  tdm_g0 g0;
  g0[0] = 1u;                                        // count = 1 (valid D#)
  g0[1] = lds_off;                                   // lds_addr
  g0[2] = (unsigned)(ga & 0xffffffffu);              // global_addr[31:0]
  g0[3] = (unsigned)((ga >> 32) & 0x01ffffffu) | 0x80000000u; // addr[56:32] | type=2
  tdm_g1 g1;
  g1[0] = 0x07930000u;   // data_size=3 | pad_enable | pad_interval=6 | pad_amount=3
  g1[1] = 64u << 16;     // tensor_dim0 = 64 (low 16 in bits 63:48)
  g1[2] = 64u << 16;     // tensor_dim0 hi = 0, tensor_dim1 = 64 (low 16)
  g1[3] = 64u << 16;     // tensor_dim1 hi = 0, tile_dim0 = 64
  g1[4] = 64u;           // tile_dim1 = 64, tile_dim2 = 0
  g1[5] = 128u;          // tensor_dim0_stride = 128 units (1024B)
  g1[6] = 0u;
  g1[7] = 0u;
  asm volatile("tensor_load_to_lds %0, %1" :: "s"(g0), "s"(g1) : "memory");
  __builtin_amdgcn_s_wait_tensorcnt(0);
}

// ---------------------------------------------------------------------------
// 1. Weight prep: Wt[c][k] = (bf16) W[k][c]
// ---------------------------------------------------------------------------
__global__ __launch_bounds__(256) void wprep_kernel(const float* __restrict__ w,
                                                    bf16* __restrict__ wt) {
  int idx = blockIdx.x * 256 + threadIdx.x;
  int c  = idx >> 9;
  int kk = idx & 511;
  wt[(size_t)c * D_ + kk] = (bf16)w[(size_t)kk * D_ + c];
}

// ---------------------------------------------------------------------------
// 2. Adjacency bit-pack
// ---------------------------------------------------------------------------
__global__ __launch_bounds__(256) void maskpack_kernel(const int* __restrict__ adj,
                                                       unsigned* __restrict__ mask) {
  size_t idx = (size_t)blockIdx.x * 256 + threadIdx.x;
  if (idx >= (size_t)B_ * N_ * NW_) return;
  const int* p = adj + idx * 32;
  unsigned w = 0;
#pragma unroll
  for (int i = 0; i < 32; ++i) w |= (p[i] != 0 ? 1u : 0u) << i;
  mask[idx] = w;
}

// ---------------------------------------------------------------------------
// 3. QKV projection.  grid = (128, 8, 3), block = 128 (4 waves).
//    Weight tile (64 cols x 512 K) TDM-staged to LDS in two 256-K phases.
// ---------------------------------------------------------------------------
__global__ __launch_bounds__(128) void qkv_kernel(
    const float* __restrict__ X,
    const bf16* __restrict__ wqt, const bf16* __restrict__ wkt, const bf16* __restrict__ wvt,
    const float* __restrict__ bq, const float* __restrict__ bk, const float* __restrict__ bv,
    bf16* __restrict__ qout, bf16* __restrict__ kout, bf16* __restrict__ vtout) {
  __shared__ bf16 wtile[64 * WROW_];
  int lane = threadIdx.x & 31;
  int wid  = threadIdx.x >> 5;
  int rb   = blockIdx.x * 4 + wid;
  int cg   = blockIdx.y;
  int which = blockIdx.z;

  const bf16*  Wt   = (which == 0) ? wqt : (which == 1) ? wkt : wvt;
  const float* bias = (which == 0) ? bq  : (which == 1) ? bk  : bv;

  int colL = lane & 15;
  const float* xrow = X + (size_t)(rb * 16 + colL) * D_;
  unsigned lds_off = (unsigned)(uintptr_t)&wtile[0];
  const bf16* wtbase = Wt + (size_t)cg * 64 * D_;

  v8f acc[4] = {};
#pragma unroll
  for (int p = 0; p < 2; ++p) {
    __syncthreads();                       // phase p-1 consumers done
    if (wid == 0)
      tdm_load_wtile(lds_off, wtbase + p * 256);   // +256 K elems (512B)
    __syncthreads();                       // tile visible to all waves

    v16bf a = load_a_f32(xrow, p * 256, lane);
#pragma unroll
    for (int k0 = 0; k0 < 256; k0 += 32) {
      v16bf an = a;
      if (k0 < 224) an = load_a_f32(xrow, p * 256 + k0 + 32, lane);  // prefetch
#pragma unroll
      for (int j = 0; j < 4; ++j) {
        const bf16* cb = &wtile[(j * 16 + colL) * WROW_];
        v16bf b = load_b_bf16(cb, k0, lane);
        acc[j] = wmma_bf16(a, b, acc[j]);
      }
      a = an;
    }
  }

  // epilogue: +bias, bf16 convert, scatter into attention-friendly layouts
#pragma unroll
  for (int j = 0; j < 4; ++j) {
    int col = cg * 64 + j * 16 + colL;
    int h   = col >> 6;
    int hd  = col & 63;
    float bb = bias[col];
#pragma unroll
    for (int r = 0; r < 8; ++r) {
      int rowg = rb * 16 + ((lane & 16) ? r + 8 : r);
      int bidx = rowg >> 11;
      int n    = rowg & (N_ - 1);
      bf16 o = (bf16)(acc[j][r] + bb);
      if (which == 0)
        qout[(((size_t)bidx * H_ + h) * N_ + n) * HD_ + hd] = o;
      else if (which == 1)
        kout[(((size_t)bidx * H_ + h) * N_ + n) * HD_ + hd] = o;
      else
        vtout[(((size_t)bidx * H_ + h) * HD_ + hd) * N_ + n] = o;
    }
  }
}

// ---------------------------------------------------------------------------
// 4. Flash attention.  grid = 1024, block = 128 (4 waves, consecutive qb
//    of the same (b,h)).  Chunk = 32 keys, K/V fragments prefetched.
// ---------------------------------------------------------------------------
__global__ __launch_bounds__(128) void attn_kernel(
    const bf16* __restrict__ q, const bf16* __restrict__ k,
    const bf16* __restrict__ vt, const unsigned* __restrict__ maskbits,
    bf16* __restrict__ attout) {
  __shared__ bf16 Ps[4][16 * 40];
  int lane = threadIdx.x & 31;
  int wid  = threadIdx.x >> 5;
  int blk  = blockIdx.x * 4 + wid;
  int qb = blk & 127;
  int h  = (blk >> 7) & 7;
  int b  = blk >> 10;
  int colL = lane & 15;

  const bf16* Qbase = q  + (((size_t)b * H_ + h) * N_ + qb * 16) * HD_;
  const bf16* Kbase = k  + (((size_t)b * H_ + h) * N_) * HD_;
  const bf16* Vt    = vt + (((size_t)b * H_ + h) * HD_) * N_;
  const unsigned* mbits = maskbits + ((size_t)b * N_ + qb * 16) * NW_;

  const bf16* qrow = Qbase + (size_t)colL * HD_;
  v16bf aQ0 = load_a_bf16(qrow, 0, lane);
  v16bf aQ1 = load_a_bf16(qrow, 32, lane);

  v8f O[4] = {};
  float mrun[8], lrun[8];
#pragma unroll
  for (int r = 0; r < 8; ++r) { mrun[r] = -3.0e38f; lrun[r] = 0.0f; }

  bf16* ps = &Ps[wid][0];
  const float scale = 0.125f;

  v16bf bK[4];
  auto loadK = [&](int c0) {
    const bf16* kr0 = Kbase + (size_t)(c0 + colL) * HD_;
    bK[0] = load_b_bf16(kr0, 0, lane);
    bK[1] = load_b_bf16(kr0, 32, lane);
    const bf16* kr1 = Kbase + (size_t)(c0 + 16 + colL) * HD_;
    bK[2] = load_b_bf16(kr1, 0, lane);
    bK[3] = load_b_bf16(kr1, 32, lane);
  };
  loadK(0);

  for (int c0 = 0; c0 < N_; c0 += 32) {
    // ---- scores from prefetched K fragments ----
    v8f s0 = {}, s1 = {};
    s0 = wmma_bf16(aQ0, bK[0], s0);
    s0 = wmma_bf16(aQ1, bK[1], s0);
    s1 = wmma_bf16(aQ0, bK[2], s1);
    s1 = wmma_bf16(aQ1, bK[3], s1);

    // ---- prefetch V fragments (overlap with softmax VALU below) ----
    v16bf bV[4];
#pragma unroll
    for (int j = 0; j < 4; ++j)
      bV[j] = load_b_bf16(Vt + (size_t)(j * 16 + colL) * N_, c0, lane);

    // ---- scale + adjacency mask ----
    unsigned mw[8];
#pragma unroll
    for (int r = 0; r < 8; ++r) {
      int rloc = (lane & 16) ? r + 8 : r;
      mw[r] = mbits[(size_t)rloc * NW_ + (c0 >> 5)];
    }
#pragma unroll
    for (int r = 0; r < 8; ++r) {
      float v0 = s0[r] * scale;
      float v1 = s1[r] * scale;
      if (!((mw[r] >> colL) & 1u))        v0 = -1.0e9f;
      if (!((mw[r] >> (16 + colL)) & 1u)) v1 = -1.0e9f;
      s0[r] = v0; s1[r] = v1;
    }

    // ---- online softmax update (rows live in 16-lane halves) ----
#pragma unroll
    for (int r = 0; r < 8; ++r) {
      float t = fmaxf(s0[r], s1[r]);
      t = fmaxf(t, __shfl_xor(t, 1, 32));
      t = fmaxf(t, __shfl_xor(t, 2, 32));
      t = fmaxf(t, __shfl_xor(t, 4, 32));
      t = fmaxf(t, __shfl_xor(t, 8, 32));
      float mnew  = fmaxf(mrun[r], t);
      float alpha = __expf(mrun[r] - mnew);
      float p0 = __expf(s0[r] - mnew);
      float p1 = __expf(s1[r] - mnew);
      s0[r] = p0; s1[r] = p1;
      float rs = p0 + p1;
      rs += __shfl_xor(rs, 1, 32);
      rs += __shfl_xor(rs, 2, 32);
      rs += __shfl_xor(rs, 4, 32);
      rs += __shfl_xor(rs, 8, 32);
      lrun[r] = lrun[r] * alpha + rs;
      mrun[r] = mnew;
#pragma unroll
      for (int j = 0; j < 4; ++j) O[j][r] = O[j][r] * alpha;
    }

    // ---- P: C-layout -> LDS -> A-layout ----
#pragma unroll
    for (int r = 0; r < 8; ++r) {
      int rloc = (lane & 16) ? r + 8 : r;
      ps[rloc * 40 + colL]      = (bf16)s0[r];
      ps[rloc * 40 + 16 + colL] = (bf16)s1[r];
    }
    asm volatile("s_wait_dscnt 0" ::: "memory");   // cross-lane LDS RAW
    v16bf aP;
    {
      int kb = (lane & 16) ? 8 : 0;
      union { v16bf v; v8bf hh[2]; } u;
      u.hh[0] = *(const v8bf*)(ps + colL * 40 + kb);
      u.hh[1] = *(const v8bf*)(ps + colL * 40 + 16 + kb);
      aP = u.v;
    }

    // ---- prefetch next chunk's K (overlaps P*V WMMAs) ----
    if (c0 + 32 < N_) loadK(c0 + 32);

    // ---- O += P * V ----
#pragma unroll
    for (int j = 0; j < 4; ++j) O[j] = wmma_bf16(aP, bV[j], O[j]);
  }

  // ---- normalize + store attended (bf16, [B*N, D]) ----
#pragma unroll
  for (int r = 0; r < 8; ++r) {
    float inv = 1.0f / lrun[r];
    int rloc = (lane & 16) ? r + 8 : r;
    size_t g = (size_t)b * N_ + qb * 16 + rloc;
#pragma unroll
    for (int j = 0; j < 4; ++j)
      attout[g * D_ + h * HD_ + j * 16 + colL] = (bf16)(O[j][r] * inv);
  }
}

// ---------------------------------------------------------------------------
// 5. Output projection + bias + residual.  grid = (128, 8), block = 128.
// ---------------------------------------------------------------------------
__global__ __launch_bounds__(128) void proj_kernel(
    const bf16* __restrict__ att, const bf16* __restrict__ wot,
    const float* __restrict__ bo, const float* __restrict__ X,
    float* __restrict__ y) {
  __shared__ bf16 wtile[64 * WROW_];
  int lane = threadIdx.x & 31;
  int wid  = threadIdx.x >> 5;
  int rb   = blockIdx.x * 4 + wid;
  int cg   = blockIdx.y;
  int colL = lane & 15;

  const bf16* arow = att + (size_t)(rb * 16 + colL) * D_;
  unsigned lds_off = (unsigned)(uintptr_t)&wtile[0];
  const bf16* wtbase = wot + (size_t)cg * 64 * D_;

  v8f acc[4] = {};
#pragma unroll
  for (int p = 0; p < 2; ++p) {
    __syncthreads();
    if (wid == 0)
      tdm_load_wtile(lds_off, wtbase + p * 256);
    __syncthreads();

    v16bf a = load_a_bf16(arow, p * 256, lane);
#pragma unroll
    for (int k0 = 0; k0 < 256; k0 += 32) {
      v16bf an = a;
      if (k0 < 224) an = load_a_bf16(arow, p * 256 + k0 + 32, lane);
#pragma unroll
      for (int j = 0; j < 4; ++j) {
        const bf16* cb = &wtile[(j * 16 + colL) * WROW_];
        v16bf b = load_b_bf16(cb, k0, lane);
        acc[j] = wmma_bf16(a, b, acc[j]);
      }
      a = an;
    }
  }
#pragma unroll
  for (int j = 0; j < 4; ++j) {
    int col = cg * 64 + j * 16 + colL;
    float bb = bo[col];
#pragma unroll
    for (int r = 0; r < 8; ++r) {
      int rowg = rb * 16 + ((lane & 16) ? r + 8 : r);
      y[(size_t)rowg * D_ + col] = acc[j][r] + bb + X[(size_t)rowg * D_ + col];
    }
  }
}

// ---------------------------------------------------------------------------
// 6. LayerNorm.  One wave per row; grid = 2048, block = 128.
// ---------------------------------------------------------------------------
__global__ __launch_bounds__(128) void ln_kernel(
    const float* __restrict__ y, const float* __restrict__ gamma,
    const float* __restrict__ beta, float* __restrict__ out) {
  int lane = threadIdx.x & 31;
  int wid  = threadIdx.x >> 5;
  int row  = blockIdx.x * 4 + wid;
  const float* p = y + (size_t)row * D_;

  v4f v[4];
  float s = 0.0f, s2 = 0.0f;
#pragma unroll
  for (int i = 0; i < 4; ++i) {
    v[i] = *(const v4f*)(p + i * 128 + lane * 4);
#pragma unroll
    for (int c = 0; c < 4; ++c) { s += v[i][c]; s2 += v[i][c] * v[i][c]; }
  }
  s  += __shfl_xor(s, 1, 32);  s2 += __shfl_xor(s2, 1, 32);
  s  += __shfl_xor(s, 2, 32);  s2 += __shfl_xor(s2, 2, 32);
  s  += __shfl_xor(s, 4, 32);  s2 += __shfl_xor(s2, 4, 32);
  s  += __shfl_xor(s, 8, 32);  s2 += __shfl_xor(s2, 8, 32);
  s  += __shfl_xor(s, 16, 32); s2 += __shfl_xor(s2, 16, 32);
  float mu  = s * (1.0f / D_);
  float var = s2 * (1.0f / D_) - mu * mu;
  float inv = rsqrtf(var + 1e-5f);
#pragma unroll
  for (int i = 0; i < 4; ++i) {
#pragma unroll
    for (int c = 0; c < 4; ++c) {
      int idx = i * 128 + lane * 4 + c;
      out[(size_t)row * D_ + idx] = (v[i][c] - mu) * inv * gamma[idx] + beta[idx];
    }
  }
}

// ---------------------------------------------------------------------------
// Launch
// ---------------------------------------------------------------------------
extern "C" void kernel_launch(void* const* d_in, const int* in_sizes, int n_in,
                              void* d_out, int out_size, void* d_ws, size_t ws_size,
                              hipStream_t stream) {
  (void)in_sizes; (void)n_in; (void)out_size; (void)ws_size;

  const float* X    = (const float*)d_in[0];
  const int*   adj  = (const int*)d_in[3];
  const float* Wq   = (const float*)d_in[5];
  const float* bq   = (const float*)d_in[6];
  const float* Wk   = (const float*)d_in[7];
  const float* bk   = (const float*)d_in[8];
  const float* Wv   = (const float*)d_in[9];
  const float* bv   = (const float*)d_in[10];
  const float* Wo   = (const float*)d_in[11];
  const float* bo   = (const float*)d_in[12];
  const float* gam  = (const float*)d_in[13];
  const float* bet  = (const float*)d_in[14];
  float* out = (float*)d_out;

  char* ws = (char*)d_ws;                       // needs >= 52 MiB
  bf16*     wqt  = (bf16*)(ws + OFF_WQT);
  bf16*     wkt  = (bf16*)(ws + OFF_WKT);
  bf16*     wvt  = (bf16*)(ws + OFF_WVT);
  bf16*     wot  = (bf16*)(ws + OFF_WOT);
  bf16*     qbuf = (bf16*)(ws + OFF_Q);
  bf16*     kbuf = (bf16*)(ws + OFF_K);
  bf16*     vtb  = (bf16*)(ws + OFF_VT);
  unsigned* mb   = (unsigned*)(ws + OFF_MASK);
  bf16*     att  = (bf16*)(ws + OFF_ATT);
  float*    ybuf = (float*)(ws + OFF_Y);

  wprep_kernel<<<1024, 256, 0, stream>>>(Wq, wqt);
  wprep_kernel<<<1024, 256, 0, stream>>>(Wk, wkt);
  wprep_kernel<<<1024, 256, 0, stream>>>(Wv, wvt);
  wprep_kernel<<<1024, 256, 0, stream>>>(Wo, wot);

  maskpack_kernel<<<2048, 256, 0, stream>>>(adj, mb);

  qkv_kernel<<<dim3(128, 8, 3), 128, 0, stream>>>(
      X, wqt, wkt, wvt, bq, bk, bv, qbuf, kbuf, vtb);

  attn_kernel<<<1024, 128, 0, stream>>>(qbuf, kbuf, vtb, mb, att);

  proj_kernel<<<dim3(128, 8), 128, 0, stream>>>(att, wot, bo, X, ybuf);

  ln_kernel<<<2048, 128, 0, stream>>>(ybuf, gam, bet, out);
}
